// AttentionBlock_25048249270569
// MI455X (gfx1250) — matrix operand order repaired and checked by reading, not compiled, not run on hardware
//
#include <hip/hip_runtime.h>

// ---------------------------------------------------------------------------
// AttentionBlock for MI455X (gfx1250, wave32, WMMA + TDM).
//   B=8, L=1024, C=1024, H=8, ch=128.
// Pipeline: LN -> QKV GEMM (wmma f16, TDM-staged A tiles) -> flash attention
//           (wmma f16) -> out GEMM + bias + residual (f32 accum everywhere).
// The reference's raw .view reshapes make flat [B,L,C] == flat [B*H,L,ch],
// so q/k/v/attn are kept as flat f16 buffers with no transposes.
// ---------------------------------------------------------------------------

#define BB   8
#define LL   1024
#define CC   1024
#define HH   8
#define CHD  128                 // CC / HH
#define ROWS (BB * LL)           // 8192
#define NHEAD (BB * HH)          // 64
#define EPSV 1e-5f

typedef __attribute__((ext_vector_type(16))) _Float16 v16h;
typedef __attribute__((ext_vector_type(8)))  _Float16 v8h;
typedef __attribute__((ext_vector_type(8)))  float    v8f;
typedef __attribute__((ext_vector_type(4)))  unsigned int v4u;
typedef __attribute__((ext_vector_type(4)))  int v4i;
typedef __attribute__((ext_vector_type(8)))  int v8i;

#if defined(__gfx1250__) && __has_builtin(__builtin_amdgcn_tensor_load_to_lds)
#define USE_TDM 1
#else
#define USE_TDM 0
#endif

static __device__ __forceinline__ v8f wmma_f16(v16h a, v16h b, v8f c) {
  // D = A(16x32 f16) * B(32x16 f16) + C(16x16 f32)
  return __builtin_amdgcn_wmma_f32_16x16x32_f16(
      /*neg_a=*/false, a, /*neg_b=*/false, b,
      /*c_mod=*/(short)0, c, /*reuse_a=*/false, /*reuse_b=*/false);
}

#if USE_TDM
// 2D f16 tile DMA: global (row-major, row_stride elems) -> LDS (contiguous
// tile_w*2-byte rows).  D# packing per CDNA5 ISA.
static __device__ __forceinline__ void tdm_load_2d_f16(
    unsigned lds_off, const void* gptr, unsigned tile_w, unsigned tile_h,
    unsigned tensor_w, unsigned tensor_h, unsigned row_stride_elems) {
  unsigned long long ga = (unsigned long long)(size_t)gptr;
  v4u g0;
  g0[0] = 1u;                                                // count=1 (valid)
  g0[1] = lds_off;                                           // lds_addr (bytes)
  g0[2] = (unsigned)(ga & 0xffffffffu);                      // global_addr lo
  g0[3] = (unsigned)((ga >> 32) & 0x01ffffffu) | 0x80000000u; // addr hi | type=2
  v8i g1;
  g1[0] = (int)(1u << 16);                                   // data_size=1 -> 2B
  g1[1] = (int)((tensor_w & 0xffffu) << 16);                 // tensor_dim0[15:0]
  g1[2] = (int)((tensor_w >> 16) | ((tensor_h & 0xffffu) << 16));
  g1[3] = (int)((tensor_h >> 16) | (tile_w << 16));          // tile_dim0
  g1[4] = (int)(tile_h & 0xffffu);                           // tile_dim1 (dim2=0)
  g1[5] = (int)row_stride_elems;                             // dim0_stride lo
  g1[6] = 0;                                                 // stride hi / dim1_stride
  g1[7] = 0;
  v4i z4 = {0, 0, 0, 0};                                     // 2D: groups 2/3 unused
#if __clang_major__ >= 23
  v8i z8 = {0, 0, 0, 0, 0, 0, 0, 0};
  __builtin_amdgcn_tensor_load_to_lds(g0, g1, z4, z4, z8, 0);
#else
  __builtin_amdgcn_tensor_load_to_lds(g0, g1, z4, z4, 0);
#endif
}
#endif

// ---------------------------------------------------------------------------
// LayerNorm over C=1024; one block (256 threads) per row.
// ---------------------------------------------------------------------------
__global__ __launch_bounds__(256)
void ln_kernel(const float* __restrict__ x, const float* __restrict__ g,
               const float* __restrict__ b, float* __restrict__ xn,
               _Float16* __restrict__ xnh) {
  const int row = blockIdx.x;
  const int t   = threadIdx.x;
  const float* xr = x + (size_t)row * CC;

  float v[4];
  float s = 0.f;
#pragma unroll
  for (int i = 0; i < 4; ++i) { v[i] = xr[t + i * 256]; s += v[i]; }

  __shared__ float red[8];
  for (int m = 16; m >= 1; m >>= 1) s += __shfl_xor(s, m, 32);
  if ((t & 31) == 0) red[t >> 5] = s;
  __syncthreads();
  float tot = 0.f;
#pragma unroll
  for (int i = 0; i < 8; ++i) tot += red[i];
  const float mu = tot * (1.0f / CC);
  __syncthreads();

  float s2 = 0.f;
#pragma unroll
  for (int i = 0; i < 4; ++i) { float d = v[i] - mu; s2 += d * d; }
  for (int m = 16; m >= 1; m >>= 1) s2 += __shfl_xor(s2, m, 32);
  if ((t & 31) == 0) red[t >> 5] = s2;
  __syncthreads();
  float tot2 = 0.f;
#pragma unroll
  for (int i = 0; i < 8; ++i) tot2 += red[i];
  const float rstd = rsqrtf(tot2 * (1.0f / CC) + EPSV);

#pragma unroll
  for (int i = 0; i < 4; ++i) {
    int c = t + i * 256;
    float y = (v[i] - mu) * rstd * g[c] + b[c];
    xn[(size_t)row * CC + c]  = y;
    xnh[(size_t)row * CC + c] = (_Float16)y;
  }
}

// ---------------------------------------------------------------------------
// f32 -> f16 elementwise (weight conversion)
// ---------------------------------------------------------------------------
__global__ __launch_bounds__(256)
void cvt_kernel(const float* __restrict__ in, _Float16* __restrict__ out, int n) {
  int i = blockIdx.x * 256 + threadIdx.x;
  if (i < n) out[i] = (_Float16)in[i];
}

// ---------------------------------------------------------------------------
// Tiled WMMA GEMM:  Cmat[M][N] = A[M][K] * Bw[K][N]  (f16 in, f32 accum)
// Block: 256 threads = 8 waves, tile 128(M) x 128(N), K step 32.
// A tiles are TDM-DMA'd into double-buffered LDS (wave 0 issues, tensorcnt
// sync); B tiles are staged transposed so fragments are contiguous LDS loads.
// mode 0: epilogue splits columns into q (*scale), k (*scale), v  (f16 out)
// mode 1: epilogue adds bias + residual, writes f32 out
// ---------------------------------------------------------------------------
__global__ __launch_bounds__(256)
void gemm_kernel(const _Float16* __restrict__ A, const _Float16* __restrict__ Bw,
                 const float* __restrict__ bias, int M, int N, int K,
                 int mode, float qkscale,
                 _Float16* __restrict__ qo, _Float16* __restrict__ ko,
                 _Float16* __restrict__ vo,
                 const float* __restrict__ resid, float* __restrict__ outf) {
  __shared__ __align__(32) _Float16 At[2][128][32];  // double-buffered A tile
  __shared__ __align__(32) _Float16 Bt[128][32];     // B tile TRANSPOSED: Bt[n][k]

  const int t    = threadIdx.x;
  const int lane = t & 31;
  const int wave = t >> 5;
  const int wm   = wave & 3;     // 0..3 : rows wm*32
  const int wn   = wave >> 2;    // 0..1 : cols wn*64
  const int lh   = lane >> 4;    // half-wave 0/1
  const int ll   = lane & 15;
  const int row0 = blockIdx.y * 128;
  const int col0 = blockIdx.x * 128;

  v8f acc[2][4];
#pragma unroll
  for (int i = 0; i < 2; ++i)
#pragma unroll
    for (int j = 0; j < 4; ++j)
#pragma unroll
      for (int r = 0; r < 8; ++r) acc[i][j][r] = 0.f;

  const int lin = t * 16;
  const int ar_ = lin >> 5;        // A tile row (0..127)  [fallback path]
  const int ac_ = lin & 31;        // 0 or 16              [fallback path]
  const int bk_ = lin >> 7;        // B tile k row (0..31)
  const int bn_ = lin & 127;       // B tile col base (multiple of 16)

#if USE_TDM
  const unsigned atbase = (unsigned)(size_t)&At[0][0][0];
  if (wave == 0)   // prologue: DMA tile 0 into buffer 0
    tdm_load_2d_f16(atbase, A + (size_t)row0 * K, 32, 128,
                    (unsigned)K, (unsigned)M, (unsigned)K);
#endif

  int cur = 0;
  for (int k0 = 0; k0 < K; k0 += 32) {
#if USE_TDM
    if (wave == 0) {
      // issue next tile into the other buffer (last iter: harmless refetch
      // so the tensorcnt wait below stays a constant immediate)
      const int kn = (k0 + 32 < K) ? (k0 + 32) : k0;
      tdm_load_2d_f16(atbase + (unsigned)((cur ^ 1) * 128 * 32 * 2),
                      A + (size_t)row0 * K + kn, 32, 128,
                      (unsigned)K, (unsigned)M, (unsigned)K);
    }
#else
    *(v16h*)&At[cur][ar_][ac_] =
        *(const v16h*)(A + (size_t)(row0 + ar_) * K + k0 + ac_);
#endif
    // --- stage B tile transposed ---
    v16h bv = *(const v16h*)(Bw + (size_t)(k0 + bk_) * N + col0 + bn_);
#pragma unroll
    for (int j = 0; j < 16; ++j) Bt[bn_ + j][bk_] = bv[j];

#if USE_TDM
    // in-order TDM: <=1 outstanding means tile `cur` has landed in LDS
    if (wave == 0) __builtin_amdgcn_s_wait_tensorcnt((short)1);
#endif
    __syncthreads();

#pragma unroll
    for (int mt = 0; mt < 2; ++mt) {
      // A fragment, rows wm*32 + mt*16 (contiguous k order: k = lh*16 + j, as B)
      const _Float16* arp = &At[cur][wm * 32 + mt * 16 + ll][0];
      v8h a0 = *(const v8h*)(arp + lh * 16);
      v8h a1 = *(const v8h*)(arp + lh * 16 + 8);
      v16h af;
#pragma unroll
      for (int j = 0; j < 8; ++j) { af[j] = a0[j]; af[8 + j] = a1[j]; }
#pragma unroll
      for (int nt = 0; nt < 4; ++nt) {
        // B fragment from transposed tile: contiguous (32x16 f16 B layout)
        v16h bf = *(const v16h*)&Bt[wn * 64 + nt * 16 + ll][lh * 16];
        acc[mt][nt] = wmma_f16(af, bf, acc[mt][nt]);
      }
    }
    __syncthreads();   // all waves done with At[cur] / Bt before rewrite
    cur ^= 1;
  }

  // --- epilogue ---
#pragma unroll
  for (int mt = 0; mt < 2; ++mt)
#pragma unroll
    for (int nt = 0; nt < 4; ++nt)
#pragma unroll
      for (int r = 0; r < 8; ++r) {
        const int m    = wm * 32 + mt * 16 + lh * 8 + r;  // C/D: M = r + 8*lh
        const int n    = wn * 64 + nt * 16 + ll;
        const int grow = row0 + m;
        const int gcol = col0 + n;
        float val = acc[mt][nt][r] + bias[gcol];
        if (mode == 0) {
          if (gcol < 1024)
            qo[(size_t)grow * 1024 + gcol] = (_Float16)(val * qkscale);
          else if (gcol < 2048)
            ko[(size_t)grow * 1024 + (gcol - 1024)] = (_Float16)(val * qkscale);
          else
            vo[(size_t)grow * 1024 + (gcol - 2048)] = (_Float16)val;
        } else {
          outf[(size_t)grow * 1024 + gcol] =
              val + resid[(size_t)grow * 1024 + gcol];
        }
      }
}

// ---------------------------------------------------------------------------
// Flash attention: Q,K,V,O are flat f16 [NHEAD][L][CHD].
// Block: 128 threads = 4 waves, each wave owns 16 query rows (64/block).
// Per 32-key step: 8 WMMAs for S, online softmax, 8 WMMAs for P*V.
// ---------------------------------------------------------------------------
__global__ __launch_bounds__(128)
void attn_kernel(const _Float16* __restrict__ Q, const _Float16* __restrict__ Kp,
                 const _Float16* __restrict__ V, _Float16* __restrict__ O) {
  __shared__ __align__(32) _Float16 Ks[32][128];      // K tile   [key][ch]
  __shared__ __align__(32) _Float16 Vt[128][32];      // V tile T [ch][key]
  __shared__ __align__(32) _Float16 Ps[4][16][32];    // per-wave P [m][k]

  const int t    = threadIdx.x;
  const int lane = t & 31;
  const int wave = t >> 5;
  const int lh   = lane >> 4;
  const int ll   = lane & 15;
  const int head = blockIdx.y;
  const int q0   = blockIdx.x * 64 + wave * 16;

  const _Float16* Qh = Q + (size_t)head * LL * CHD;
  const _Float16* Kh = Kp + (size_t)head * LL * CHD;
  const _Float16* Vh = V + (size_t)head * LL * CHD;

  // preload this wave's Q fragments (16 rows x 128 ch = 4 chunks of K=32)
  v16h qf[4];
#pragma unroll
  for (int c = 0; c < 4; ++c) {
    const _Float16* qr = Qh + (size_t)(q0 + ll) * CHD + c * 32;
    v8h a0 = *(const v8h*)(qr + lh * 16);
    v8h a1 = *(const v8h*)(qr + lh * 16 + 8);
#pragma unroll
    for (int j = 0; j < 8; ++j) { qf[c][j] = a0[j]; qf[c][8 + j] = a1[j]; }
  }

  v8f o[8];
#pragma unroll
  for (int cb = 0; cb < 8; ++cb)
#pragma unroll
    for (int r = 0; r < 8; ++r) o[cb][r] = 0.f;

  float rm[8], rs[8];
#pragma unroll
  for (int r = 0; r < 8; ++r) { rm[r] = -1e30f; rs[r] = 0.f; }

  const int lin = t * 32;          // cooperative staging: 32 halves/thread
  const int kr_ = lin >> 7;        // key row 0..31
  const int cc_ = lin & 127;       // 0,32,64,96

  for (int kb = 0; kb < LL; kb += 32) {
    __syncthreads();   // previous iteration's tile reads complete
    // --- stage K tile and transposed V tile ---
    {
      const _Float16* ksrc = Kh + (size_t)(kb + kr_) * CHD + cc_;
      *(v16h*)&Ks[kr_][cc_]      = *(const v16h*)ksrc;
      *(v16h*)&Ks[kr_][cc_ + 16] = *(const v16h*)(ksrc + 16);
      const _Float16* vsrc = Vh + (size_t)(kb + kr_) * CHD + cc_;
      v16h v0 = *(const v16h*)vsrc;
      v16h v1 = *(const v16h*)(vsrc + 16);
#pragma unroll
      for (int j = 0; j < 16; ++j) {
        Vt[cc_ + j][kr_]      = v0[j];
        Vt[cc_ + 16 + j][kr_] = v1[j];
      }
    }
    __syncthreads();

    // --- S = Q * K^T : two 16x16 f32 tiles (keys 0-15 / 16-31) ---
    v8f s0, s1;
#pragma unroll
    for (int r = 0; r < 8; ++r) { s0[r] = 0.f; s1[r] = 0.f; }
#pragma unroll
    for (int c = 0; c < 4; ++c) {
      v16h b0 = *(const v16h*)&Ks[ll][c * 32 + lh * 16];
      v16h b1 = *(const v16h*)&Ks[16 + ll][c * 32 + lh * 16];
      s0 = wmma_f16(qf[c], b0, s0);
      s1 = wmma_f16(qf[c], b1, s1);
    }

    // --- online softmax (row stats live in the 16-lane half owning each row)
#pragma unroll
    for (int r = 0; r < 8; ++r) {
      float v0 = s0[r], v1 = s1[r];
      float mx = fmaxf(v0, v1);
      for (int m = 1; m <= 8; m <<= 1) mx = fmaxf(mx, __shfl_xor(mx, m, 32));
      float nm = fmaxf(rm[r], mx);
      float p0 = __expf(v0 - nm);
      float p1 = __expf(v1 - nm);
      float ps = p0 + p1;
      for (int m = 1; m <= 8; m <<= 1) ps += __shfl_xor(ps, m, 32);
      float sc = __expf(rm[r] - nm);
      rs[r] = rs[r] * sc + ps;
      rm[r] = nm;
#pragma unroll
      for (int cb = 0; cb < 8; ++cb) o[cb][r] *= sc;
      const int mrow = lh * 8 + r;                 // D layout -> row index
      Ps[wave][mrow][ll]      = (_Float16)p0;
      Ps[wave][mrow][16 + ll] = (_Float16)p1;
    }
    __syncthreads();   // P (D layout) -> A layout via LDS

    // --- O += P * V ---
    const _Float16* prp = &Ps[wave][ll][0];
    v8h a0 = *(const v8h*)(prp + lh * 16);
    v8h a1 = *(const v8h*)(prp + lh * 16 + 8);
    v16h pf;
#pragma unroll
    for (int j = 0; j < 8; ++j) { pf[j] = a0[j]; pf[8 + j] = a1[j]; }
#pragma unroll
    for (int cb = 0; cb < 8; ++cb) {
      v16h bf = *(const v16h*)&Vt[cb * 16 + ll][lh * 16];
      o[cb] = wmma_f16(pf, bf, o[cb]);
    }
  }

  // --- finalize: divide by row sums, store f16 ---
#pragma unroll
  for (int cb = 0; cb < 8; ++cb)
#pragma unroll
    for (int r = 0; r < 8; ++r) {
      const int mrow = lh * 8 + r;
      float val = o[cb][r] / rs[r];
      O[(size_t)head * LL * CHD + (size_t)(q0 + mrow) * CHD + cb * 16 + ll] =
          (_Float16)val;
    }
}

// ---------------------------------------------------------------------------
// Host launcher. Workspace layout (~126 MB, 256B-aligned carve):
//   xn_f32[ROWS*C] | xn_f16 | wqkv_f16[1024*3072] | wout_f16[1024*1024]
//   | q_f16 | k_f16 | v_f16 | attn_f16   (each ROWS*C halves)
// ---------------------------------------------------------------------------
extern "C" void kernel_launch(void* const* d_in, const int* in_sizes, int n_in,
                              void* d_out, int out_size, void* d_ws, size_t ws_size,
                              hipStream_t stream) {
  (void)in_sizes; (void)n_in; (void)out_size; (void)ws_size;

  const float* x     = (const float*)d_in[0];
  const float* ln_g  = (const float*)d_in[1];
  const float* ln_b  = (const float*)d_in[2];
  const float* w_qkv = (const float*)d_in[3];
  const float* b_qkv = (const float*)d_in[4];
  const float* w_out = (const float*)d_in[5];
  const float* b_out = (const float*)d_in[6];
  float* out = (float*)d_out;

  char* p = (char*)d_ws;
  auto carve = [&](size_t bytes) -> char* {
    char* r = p;
    p += (bytes + 255) & ~(size_t)255;
    return r;
  };
  float*    xn    = (float*)   carve((size_t)ROWS * CC * sizeof(float));
  _Float16* xnh   = (_Float16*)carve((size_t)ROWS * CC * sizeof(_Float16));
  _Float16* wqkvh = (_Float16*)carve((size_t)CC * 3 * CC * sizeof(_Float16));
  _Float16* wouth = (_Float16*)carve((size_t)CC * CC * sizeof(_Float16));
  _Float16* qh    = (_Float16*)carve((size_t)ROWS * CC * sizeof(_Float16));
  _Float16* kh    = (_Float16*)carve((size_t)ROWS * CC * sizeof(_Float16));
  _Float16* vh    = (_Float16*)carve((size_t)ROWS * CC * sizeof(_Float16));
  _Float16* ah    = (_Float16*)carve((size_t)ROWS * CC * sizeof(_Float16));

  const float scale = 0.29730177875068026f;  // 128^(-1/4)

  // 1) LayerNorm
  ln_kernel<<<dim3(ROWS), dim3(256), 0, stream>>>(x, ln_g, ln_b, xn, xnh);

  // 2) weight conversions
  {
    int n1 = CC * 3 * CC;
    cvt_kernel<<<dim3((n1 + 255) / 256), dim3(256), 0, stream>>>(w_qkv, wqkvh, n1);
    int n2 = CC * CC;
    cvt_kernel<<<dim3((n2 + 255) / 256), dim3(256), 0, stream>>>(w_out, wouth, n2);
  }

  // 3) QKV GEMM: [8192 x 1024] x [1024 x 3072]; epilogue scales q,k and splits
  gemm_kernel<<<dim3(3 * CC / 128, ROWS / 128), dim3(256), 0, stream>>>(
      xnh, wqkvh, b_qkv, ROWS, 3 * CC, CC, /*mode=*/0, scale,
      qh, kh, vh, nullptr, nullptr);

  // 4) flash attention: 64 heads x 16 query-blocks of 64 rows
  attn_kernel<<<dim3(LL / 64, NHEAD), dim3(128), 0, stream>>>(qh, kh, vh, ah);

  // 5) out projection + bias + residual(xn)
  gemm_kernel<<<dim3(CC / 128, ROWS / 128), dim3(256), 0, stream>>>(
      ah, wouth, b_out, ROWS, CC, CC, /*mode=*/1, 1.0f,
      nullptr, nullptr, nullptr, xn, out);
}
